// LSTMModel_36223754174704
// MI455X (gfx1250) — matrix operand order repaired and checked by reading, not compiled, not run on hardware
//
#include <hip/hip_runtime.h>
#include <hip/hip_bf16.h>

#define DI __device__ __forceinline__

typedef __bf16 v16bf __attribute__((ext_vector_type(16)));
typedef float  v8f   __attribute__((ext_vector_type(8)));
typedef unsigned int v4u __attribute__((ext_vector_type(4)));
typedef int          v8i __attribute__((ext_vector_type(8)));
typedef int          v4i __attribute__((ext_vector_type(4)));

constexpr int Bsz = 64, Seq = 512, In0 = 256, Hd = 512;
constexpr int NWG = 16, NTHR = 256;
constexpr int GATE4 = 4 * Hd;

// ---------------- workspace layout (bytes) ----------------
constexpr size_t OFF_CTRL  = 0;                                         // 2 x u32 barrier state
constexpr size_t OFF_BIAS0 = 256;                                       // [4H] f32
constexpr size_t OFF_BIAS1 = OFF_BIAS0 + (size_t)GATE4 * 4;             // [4H] f32
constexpr size_t OFF_WIH0  = (OFF_BIAS1 + (size_t)GATE4 * 4 + 255) & ~(size_t)255;
constexpr size_t OFF_WHH0  = OFF_WIH0 + (size_t)GATE4 * In0 * 2;        // bf16 [4H,H]
constexpr size_t OFF_WIH1  = OFF_WHH0 + (size_t)GATE4 * Hd  * 2;        // bf16 [4H,H]
constexpr size_t OFF_WHH1  = OFF_WIH1 + (size_t)GATE4 * Hd  * 2;        // bf16 [4H,H]
constexpr size_t OFF_XBF   = OFF_WHH1 + (size_t)GATE4 * Hd  * 2;        // bf16 [S,B,I]
constexpr size_t OFF_HS0   = OFF_XBF  + (size_t)Seq * Bsz * In0 * 2;    // bf16 [S,B,H]
constexpr size_t OFF_HB0   = OFF_HS0  + (size_t)Seq * Bsz * Hd  * 2;    // bf16 [2,B,H]
constexpr size_t OFF_HB1   = OFF_HB0  + (size_t)2 * Bsz * Hd * 2;       // bf16 [2,B,H]

// ---------------- fragment loads ----------------
struct bf16x16 { uint4 a, b; };

// B operand (K x 16): pointer offset by (lane>=16 ? 8 : 0); lane holds K {0..7,16..23} / {8..15,24..31} of one W row.
DI v16bf ldB(const __hip_bfloat16* p) {
  bf16x16 t;
  t.a = *reinterpret_cast<const uint4*>(p);
  t.b = *reinterpret_cast<const uint4*>(p + 16);
  return __builtin_bit_cast(v16bf, t);
}

// A operand (16 x 32): pointer already offset by (lane>=16 ? 8 : 0);
// lane<16 holds K {0..7,16..23}, lane>=16 holds K {8..15,24..31}.
DI v16bf ldA(const __hip_bfloat16* p) {
  bf16x16 t;
  t.a = *reinterpret_cast<const uint4*>(p);
  t.b = *reinterpret_cast<const uint4*>(p + 16);
  return __builtin_bit_cast(v16bf, t);
}

DI v8f wmma_bf16(v16bf a, v16bf b, v8f c) {
  return __builtin_amdgcn_wmma_f32_16x16x32_bf16(false, a, false, b, (short)0, c,
                                                 false, false);
}

DI float sigm_(float x) { return 1.f / (1.f + __expf(-x)); }
DI float tanh_(float x) {
  float ax = fabsf(x);
  float e  = __expf(-2.f * ax);
  float t  = (1.f - e) / (1.f + e);
  return copysignf(t, x);
}

// ---------------- Tensor Data Mover: 1-D tile global->LDS ----------------
// D# per CDNA5 ISA ch.8: group0 = {count/type/addrs}, group1 = {dims}.
// nelem = #bf16 elements (<= 65535, tile_dim0 is 16-bit). scope=DEV so the
// TDM reads L2-fresh data produced by other WGPs.
DI void tdm_load_1d(unsigned lds_off, const void* gsrc, unsigned nelem) {
  unsigned long long ga = (unsigned long long)(uintptr_t)gsrc;
  v4u g0;
  g0[0] = 1u;                                   // count=1, user descriptor
  g0[1] = lds_off;                              // lds_addr (bytes)
  g0[2] = (unsigned)ga;                         // global_addr[31:0]
  g0[3] = (unsigned)(ga >> 32) | (2u << 30);    // global_addr[56:32] | type=2
  v8i g1;
  g1[0] = (int)(1u << 16);                      // data_size=1 (2 bytes), wg_mask=0
  g1[1] = (int)((nelem & 0xFFFFu) << 16);       // tensor_dim0[15:0]
  g1[2] = (int)(((nelem >> 16) & 0xFFFFu) | (1u << 16)); // dim0 hi | tensor_dim1=1
  g1[3] = (int)((nelem & 0xFFFFu) << 16);       // tile_dim0
  g1[4] = 0;                                    // tile_dim1=0, tile_dim2=0
  g1[5] = (int)nelem;                           // tensor_dim0_stride lo
  g1[6] = 0;
  g1[7] = 0;
  v4i z4 = {0, 0, 0, 0};
  v8i z8 = {0, 0, 0, 0, 0, 0, 0, 0};
  __builtin_amdgcn_tensor_load_to_lds(g0, g1, z4, z4, z8, /*cpol: SCOPE_DEV*/ 16);
}

DI unsigned lds_off_of(const void* p) {  // flat shared addr[31:0] == LDS offset
  return (unsigned)(uintptr_t)p;
}

// ---------------- device-wide barrier (persistent kernel) ----------------
DI void grid_barrier(unsigned* ctrl) {
  __builtin_amdgcn_fence(__ATOMIC_RELEASE, "agent");   // drain h stores past WGP$
  __syncthreads();
  if (threadIdx.x == 0) {
    unsigned g = __hip_atomic_load(&ctrl[1], __ATOMIC_RELAXED, __HIP_MEMORY_SCOPE_AGENT);
    unsigned a = __hip_atomic_fetch_add(&ctrl[0], 1u, __ATOMIC_ACQ_REL, __HIP_MEMORY_SCOPE_AGENT);
    if (a == NWG - 1u) {
      __hip_atomic_store(&ctrl[0], 0u, __ATOMIC_RELAXED, __HIP_MEMORY_SCOPE_AGENT);
      __hip_atomic_fetch_add(&ctrl[1], 1u, __ATOMIC_ACQ_REL, __HIP_MEMORY_SCOPE_AGENT);
    } else {
      while (__hip_atomic_load(&ctrl[1], __ATOMIC_RELAXED, __HIP_MEMORY_SCOPE_AGENT) == g)
        __builtin_amdgcn_s_sleep(1);
    }
  }
  __syncthreads();
  __builtin_amdgcn_fence(__ATOMIC_ACQUIRE, "agent");   // invalidate L0 for fresh h
}

// ---------------- prep kernels ----------------
__global__ void prep_init(const float* __restrict__ bih0, const float* __restrict__ bhh0,
                          const float* __restrict__ bih1, const float* __restrict__ bhh1,
                          float* __restrict__ bias0, float* __restrict__ bias1,
                          __hip_bfloat16* __restrict__ hb0, __hip_bfloat16* __restrict__ hb1,
                          unsigned* __restrict__ ctrl) {
  int i = blockIdx.x * blockDim.x + threadIdx.x;
  int stride = gridDim.x * blockDim.x;
  for (int k = i; k < GATE4; k += stride) {
    bias0[k] = bih0[k] + bhh0[k];
    bias1[k] = bih1[k] + bhh1[k];
  }
  __hip_bfloat16 z = __float2bfloat16(0.f);
  for (int k = i; k < 2 * Bsz * Hd; k += stride) { hb0[k] = z; hb1[k] = z; }
  if (i == 0) { ctrl[0] = 0u; ctrl[1] = 0u; }
}

__global__ void prep_w(const float* __restrict__ w, __hip_bfloat16* __restrict__ o, int n) {
  int i = blockIdx.x * blockDim.x + threadIdx.x;
  for (int k = i; k < n; k += gridDim.x * blockDim.x) o[k] = __float2bfloat16(w[k]);
}

__global__ void prep_x(const float* __restrict__ X, __hip_bfloat16* __restrict__ o) {
  // X: [B,S,I] f32 -> o: [S,B,I] bf16 (time-major rows for A-fragments)
  int i = blockIdx.x * blockDim.x + threadIdx.x;
  int n = Bsz * Seq * In0;
  for (int k = i; k < n; k += gridDim.x * blockDim.x) {
    int ii = k % In0;
    int r  = k / In0;
    int s  = r % Seq;
    int b  = r / Seq;
    o[((size_t)s * Bsz + b) * In0 + ii] = __float2bfloat16(X[k]);
  }
}

// -------- chunk-level double-buffered LDS GEMM (A from LDS, B in VGPRs) ----
// All 8 ds_load_b128 of chunk c+1 are issued as a group before the 4 WMMAs of
// chunk c, giving a 4-WMMA issue distance between load and use (staircase
// dscnt waits instead of wait-0 per WMMA).
template <int NCH>
DI void gemm_lds(const __hip_bfloat16* arow, int ldr, const v16bf* w, v8f* acc) {
  v16bf fa[4], fb[4];
#pragma unroll
  for (int mt = 0; mt < 4; ++mt) fa[mt] = ldA(arow + (size_t)ldr * (16 * mt));
#pragma unroll
  for (int c = 0; c < NCH; ++c) {
    v16bf* cur = (c & 1) ? fb : fa;
    v16bf* nxt = (c & 1) ? fa : fb;
    if (c + 1 < NCH) {
#pragma unroll
      for (int mt = 0; mt < 4; ++mt)
        nxt[mt] = ldA(arow + (size_t)ldr * (16 * mt) + (c + 1) * 32);
    }
#pragma unroll
    for (int mt = 0; mt < 4; ++mt)
      acc[mt] = wmma_bf16(cur[mt], w[c], acc[mt]);
  }
}

// ---------------- persistent LSTM layer ----------------
// 16 WGs x 256 threads (8 waves). WG `wg` owns h columns [wg*32, wg*32+32).
// Weight B-fragments live in VGPRs for the whole sequence. A-operands are
// staged into LDS by the TDM: x_{t+1} is prefetched (double-buffered) during
// step t, and the h_t copy hides behind the input-projection GEMM via split
// s_wait_tensorcnt thresholds (TDM completes in issue order).
template <int KIN, bool LAST>
__global__ __launch_bounds__(NTHR) void lstm_layer(
    const __hip_bfloat16* __restrict__ Xin,   // [S,B,KIN] bf16
    const __hip_bfloat16* __restrict__ Wih,   // [4H,KIN] bf16 row-major
    const __hip_bfloat16* __restrict__ Whh,   // [4H,H]   bf16 row-major
    const float* __restrict__ bias,           // [4H] (b_ih + b_hh)
    __hip_bfloat16* __restrict__ hbuf,        // [2,B,H] bf16 double buffer
    __hip_bfloat16* __restrict__ hs_out,      // [S,B,H] bf16 (layer0 only)
    float* __restrict__ ht_out,               // [B,S,H] f32 (layer1 only)
    float* __restrict__ h_last,               // [B,H]
    float* __restrict__ c_last,               // [B,H]
    unsigned* __restrict__ ctrl) {
  extern __shared__ char smem[];
  // [ hstage 64KB | xbuf0 | xbuf1 | cst 8KB ]  (gacc aliases current xbuf)
  constexpr unsigned HBYTES = (unsigned)Bsz * Hd * 2;                 // 65536
  constexpr unsigned XB     = (unsigned)Bsz * KIN * 2;
  constexpr unsigned GB     = 64u * 129u * 4u;                        // 33024
  constexpr unsigned XG     = XB > GB ? XB : GB;                      // slot size
  __hip_bfloat16* hstage = (__hip_bfloat16*)smem;
  char*           xslot  = smem + HBYTES;
  float*          cst    = (float*)(smem + HBYTES + 2 * XG);

  const int wg    = blockIdx.x;
  const int j0    = wg * 32;
  const int lane  = threadIdx.x & 31;
  const int wave  = threadIdx.x >> 5;
  const int gate  = wave >> 1;
  const int sub   = wave & 1;
  const int l16   = lane & 15;
  const int ahalf = (lane >= 16) ? 8 : 0;
  const int bhalf = (lane >= 16) ? 8 : 0;
  const int wrow  = gate * Hd + j0 + sub * 16 + l16;   // W row = gate column (B op)

  for (int e = threadIdx.x; e < 64 * 32; e += NTHR) cst[e] = 0.f;

  // cache ALL weight B-fragments for this wave's 16 gate columns in VGPRs
  v16bf whh[Hd / 32];
  v16bf wih[KIN / 32];
  {
    const __hip_bfloat16* wp = Whh + (size_t)wrow * Hd + bhalf;
#pragma unroll
    for (int kc = 0; kc < Hd / 32; ++kc) whh[kc] = ldB(wp + kc * 32);
    const __hip_bfloat16* wq = Wih + (size_t)wrow * KIN + bhalf;
#pragma unroll
    for (int kc = 0; kc < KIN / 32; ++kc) wih[kc] = ldB(wq + kc * 32);
  }

  // per-thread loop-invariant state for the elementwise phase
  const int jj   = threadIdx.x & 31;
  const int jcol = j0 + jj;
  const float bi = bias[jcol];
  const float bff = bias[Hd + jcol];
  const float bg = bias[2 * Hd + jcol];
  const float bo = bias[3 * Hd + jcol];

  const unsigned hs_lds = lds_off_of(hstage);
  const unsigned xs_lds = lds_off_of(xslot);

  // prologue: prefetch x_0 into xbuf[0]
  if (wave == 0) tdm_load_1d(xs_lds, Xin, (unsigned)(Bsz * KIN));
  __syncthreads();

#pragma unroll 1
  for (int t = 0; t < Seq; ++t) {
    const unsigned cb = (unsigned)(t & 1);          // current x buffer / gacc
    const unsigned nb = cb ^ 1u;                    // prefetch target
    // ---- TDM: h_t copy + x_{t+1} prefetch (dummy x_0 on the last step) ----
    if (wave == 0) {
      tdm_load_1d(hs_lds, hbuf + (size_t)cb * Bsz * Hd, (unsigned)(Bsz * Hd));
      const int tn = (t + 1 < Seq) ? (t + 1) : 0;
      tdm_load_1d(xs_lds + nb * XG, Xin + (size_t)tn * Bsz * KIN,
                  (unsigned)(Bsz * KIN));
      __builtin_amdgcn_s_wait_tensorcnt(2);         // x_t complete (in-order)
    }
    __syncthreads();

    v8f acc[4];
#pragma unroll
    for (int mt = 0; mt < 4; ++mt)
#pragma unroll
      for (int r = 0; r < 8; ++r) acc[mt][r] = 0.f;

    // ---- input projection: x_t @ W_ih^T (h_t TDM copy still in flight) ----
    const __hip_bfloat16* xrow =
        (const __hip_bfloat16*)(xslot + cb * XG) + (size_t)l16 * KIN + ahalf;
    gemm_lds<KIN / 32>(xrow, KIN, wih, acc);

    if (wave == 0) __builtin_amdgcn_s_wait_tensorcnt(1);  // h_t complete
    __syncthreads();

    // ---- recurrent projection: h_{t-1..} @ W_hh^T ----
    const __hip_bfloat16* hrow = hstage + (size_t)l16 * Hd + ahalf;
    gemm_lds<Hd / 32>(hrow, Hd, whh, acc);

    // ---- scatter C/D tiles to LDS gacc (aliases xbuf[cb], reads finished) ----
    float* gacc = (float*)(xslot + cb * XG);
    const int colL  = wave * 16 + l16;
    const int rbase = (lane >= 16) ? 8 : 0;
#pragma unroll
    for (int mt = 0; mt < 4; ++mt)
#pragma unroll
      for (int r = 0; r < 8; ++r)
        gacc[(mt * 16 + rbase + r) * 129 + colL] = acc[mt][r];
    __syncthreads();

    // ---- gate fusion + state update (all 4 gates of a column in this WG) ----
    __hip_bfloat16* hw = hbuf + (size_t)nb * Bsz * Hd;
    for (int e = threadIdx.x; e < Bsz * 32; e += NTHR) {
      int b = e >> 5;                       // jj == e & 31 (stride 256)
      float gi = gacc[b * 129 + jj]      + bi;
      float gf = gacc[b * 129 + 32 + jj] + bff;
      float gg = gacc[b * 129 + 64 + jj] + bg;
      float go = gacc[b * 129 + 96 + jj] + bo;
      float ig = sigm_(gi), fg = sigm_(gf), g = tanh_(gg), og = sigm_(go);
      float c = fg * cst[e] + ig * g;
      cst[e] = c;
      float h = og * tanh_(c);
      hw[(size_t)b * Hd + jcol] = __float2bfloat16(h);
      if (!LAST) {
        hs_out[((size_t)t * Bsz + b) * Hd + jcol] = __float2bfloat16(h);
      } else {
        ht_out[((size_t)b * Seq + t) * Hd + jcol] = h;
      }
      if (t == Seq - 1) {
        h_last[(size_t)b * Hd + jcol] = h;
        c_last[(size_t)b * Hd + jcol] = c;
      }
    }

    grid_barrier(ctrl);   // release h_t device-wide; acquire for next step
  }
}

// ---------------- launcher ----------------
extern "C" void kernel_launch(void* const* d_in, const int* in_sizes, int n_in,
                              void* d_out, int out_size, void* d_ws, size_t ws_size,
                              hipStream_t stream) {
  const float* X    = (const float*)d_in[0];
  const float* Wih0 = (const float*)d_in[1];
  const float* Whh0 = (const float*)d_in[2];
  const float* bih0 = (const float*)d_in[3];
  const float* bhh0 = (const float*)d_in[4];
  const float* Wih1 = (const float*)d_in[5];
  const float* Whh1 = (const float*)d_in[6];
  const float* bih1 = (const float*)d_in[7];
  const float* bhh1 = (const float*)d_in[8];

  char* ws = (char*)d_ws;
  unsigned*       ctrl  = (unsigned*)(ws + OFF_CTRL);
  float*          bias0 = (float*)(ws + OFF_BIAS0);
  float*          bias1 = (float*)(ws + OFF_BIAS1);
  __hip_bfloat16* wih0  = (__hip_bfloat16*)(ws + OFF_WIH0);
  __hip_bfloat16* whh0  = (__hip_bfloat16*)(ws + OFF_WHH0);
  __hip_bfloat16* wih1  = (__hip_bfloat16*)(ws + OFF_WIH1);
  __hip_bfloat16* whh1  = (__hip_bfloat16*)(ws + OFF_WHH1);
  __hip_bfloat16* xbf   = (__hip_bfloat16*)(ws + OFF_XBF);
  __hip_bfloat16* hs0   = (__hip_bfloat16*)(ws + OFF_HS0);
  __hip_bfloat16* hb0   = (__hip_bfloat16*)(ws + OFF_HB0);
  __hip_bfloat16* hb1   = (__hip_bfloat16*)(ws + OFF_HB1);

  float* out  = (float*)d_out;
  float* ht   = out;                               // [B,S,H]
  float* allh = out + (size_t)Bsz * Seq * Hd;      // [2,B,H]
  float* allc = allh + (size_t)2 * Bsz * Hd;       // [2,B,H]

  prep_init<<<64, 256, 0, stream>>>(bih0, bhh0, bih1, bhh1, bias0, bias1, hb0, hb1, ctrl);
  prep_w<<<256, 256, 0, stream>>>(Wih0, wih0, GATE4 * In0);
  prep_w<<<256, 256, 0, stream>>>(Whh0, whh0, GATE4 * Hd);
  prep_w<<<256, 256, 0, stream>>>(Wih1, wih1, GATE4 * Hd);
  prep_w<<<256, 256, 0, stream>>>(Whh1, whh1, GATE4 * Hd);
  prep_x<<<512, 256, 0, stream>>>(X, xbf);

  // dynamic LDS: hstage + 2 x-slots (gacc aliases) + cst
  constexpr unsigned XG0 = ((unsigned)Bsz * In0 * 2 > 33024u) ? (unsigned)Bsz * In0 * 2 : 33024u;
  constexpr unsigned XG1 = ((unsigned)Bsz * Hd  * 2 > 33024u) ? (unsigned)Bsz * Hd  * 2 : 33024u;
  constexpr unsigned SM0 = 65536u + 2u * XG0 + 8192u;
  constexpr unsigned SM1 = 65536u + 2u * XG1 + 8192u;

  lstm_layer<In0, false><<<NWG, NTHR, SM0, stream>>>(
      xbf, wih0, whh0, bias0, hb0, hs0, nullptr, allh, allc, ctrl);
  lstm_layer<Hd, true><<<NWG, NTHR, SM1, stream>>>(
      hs0, wih1, whh1, bias1, hb1, nullptr, ht, allh + (size_t)Bsz * Hd,
      allc + (size_t)Bsz * Hd, ctrl);
}
